// HeteroGNN_86028194939239
// MI455X (gfx1250) — matrix operand order repaired and checked by reading, not compiled, not run on hardware
//
#include <hip/hip_runtime.h>
#include <hip/hip_bf16.h>

// ---------------- problem constants (match reference) ----------------
#define NGN 40000
#define NMH 30000
#define HD  128
#define KGE 1900
#define KME 768
#define EPP 1200000
#define EMM 800000
#define EGM 1000000

typedef __bf16 bf16;
typedef __attribute__((ext_vector_type(16))) __bf16 v16bf;
typedef __attribute__((ext_vector_type(8)))  float  v8f;

// =====================================================================
// Weight pack: float [K,128] row-major -> bf16 WMMA B-fragments.
// Fragment index = ((kt*8 + nt)*32 + lane)*16 + j
//   col = nt*16 + (lane&15);  k = kt*32 + (lane<16 ? 0 : 8) + (j<8 ? j : j+8)
// =====================================================================
__global__ __launch_bounds__(256) void pack_w128(const float* __restrict__ W,
                                                 bf16* __restrict__ Wp, int K) {
  const int Kt = (K + 31) >> 5;
  const size_t total = (size_t)Kt * 4096;
  size_t idx = (size_t)blockIdx.x * 256 + threadIdx.x;
  if (idx >= total) return;
  const int j    = (int)(idx & 15);
  const int lane = (int)((idx >> 4) & 31);
  const int nt   = (int)((idx >> 9) & 7);
  const int kt   = (int)(idx >> 12);
  const int col  = (nt << 4) + (lane & 15);
  const int k    = (kt << 5) + ((lane < 16) ? 0 : 8) + ((j < 8) ? j : j + 8);
  Wp[idx] = (k < K) ? (bf16)W[(size_t)k * HD + col] : (bf16)0.0f;
}

// A fragment (16x32 bf16, wave32 layout): lane<16 -> row M=lane, K {kb..kb+7,
// kb+16..kb+23} with kb=kt*32; lane>=16 -> same row, kb=kt*32+8.
__device__ __forceinline__ v16bf load_a_frag(const float* __restrict__ Arow,
                                             int kb, int K, bool tail) {
  v16bf a;
  if (!tail) {
    const float4 f0 = *(const float4*)(Arow + kb);
    const float4 f1 = *(const float4*)(Arow + kb + 4);
    const float4 f2 = *(const float4*)(Arow + kb + 16);
    const float4 f3 = *(const float4*)(Arow + kb + 20);
    a[0]=(bf16)f0.x; a[1]=(bf16)f0.y; a[2]=(bf16)f0.z; a[3]=(bf16)f0.w;
    a[4]=(bf16)f1.x; a[5]=(bf16)f1.y; a[6]=(bf16)f1.z; a[7]=(bf16)f1.w;
    a[8]=(bf16)f2.x; a[9]=(bf16)f2.y; a[10]=(bf16)f2.z; a[11]=(bf16)f2.w;
    a[12]=(bf16)f3.x; a[13]=(bf16)f3.y; a[14]=(bf16)f3.z; a[15]=(bf16)f3.w;
  } else {
    #pragma unroll
    for (int j = 0; j < 8; ++j) {
      const int k0 = kb + j, k1 = kb + 16 + j;
      a[j]     = (k0 < K) ? (bf16)Arow[k0] : (bf16)0.0f;
      a[j + 8] = (k1 < K) ? (bf16)Arow[k1] : (bf16)0.0f;
    }
  }
  return a;
}

// =====================================================================
// GEMM: C[M,128] = A[M,K] (f32) @ Wp (packed bf16) (+bias), f32 accum.
// Block = 128 threads = 4 waves; each wave owns a 32-row stripe and all
// 128 output columns: 2 A fragments x 8 B fragments -> 16 v_wmma per
// 32-K step. A sched_barrier pins the load phase (A + all 8 B fragments
// as one pipelined clause, single s_wait) ahead of the 16-WMMA burst.
// =====================================================================
__global__ __launch_bounds__(128) void gemm_n128(const float* __restrict__ A,
                                                 const bf16* __restrict__ Wp,
                                                 const float* __restrict__ bias,
                                                 float* __restrict__ C,
                                                 int M, int K) {
  const int lane = threadIdx.x & 31;
  const int wave = threadIdx.x >> 5;
  const int m0 = (blockIdx.x * 4 + wave) * 32;
  if (m0 >= M) return;
  const bool hi = lane >= 16;
  const int r0 = m0 + (lane & 15);
  const int r1 = r0 + 16;
  // clamp so out-of-range tiles still load in-bounds (stores are guarded)
  const float* __restrict__ Arow0 = A + (size_t)(r0 < M ? r0 : M - 1) * K;
  const float* __restrict__ Arow1 = A + (size_t)(r1 < M ? r1 : M - 1) * K;
  const int Kt = (K + 31) >> 5;

  v8f acc[16] = {};

  for (int kt = 0; kt < Kt; ++kt) {
    const bool tail = (kt << 5) + 32 > K;
    const int kb = (kt << 5) + (hi ? 8 : 0);
    const v16bf a0 = load_a_frag(Arow0, kb, K, tail);
    const v16bf a1 = load_a_frag(Arow1, kb, K, tail);

    v16bf b[8];
    const bf16* __restrict__ wb = Wp + (size_t)kt * 4096 + (size_t)lane * 16;
    #pragma unroll
    for (int nt = 0; nt < 8; ++nt)
      b[nt] = *(const v16bf*)(wb + (size_t)nt * 512);

    if (kt + 1 < Kt)                       // pull next 8KB weight chunk near
      __builtin_prefetch(wb + 4096, 0, 0); // -> global_prefetch_b8

    // keep all fragment loads above, all WMMAs below: one load clause +
    // one wait per K-step instead of a wait per fragment.
    __builtin_amdgcn_sched_barrier(0);

    #pragma unroll
    for (int nt = 0; nt < 8; ++nt) {
      acc[nt]     = __builtin_amdgcn_wmma_f32_16x16x32_bf16(
          false, a0, false, b[nt], (short)0, acc[nt], false, false);
      acc[8 + nt] = __builtin_amdgcn_wmma_f32_16x16x32_bf16(
          false, a1, false, b[nt], (short)0, acc[8 + nt], false, false);
    }
  }

  #pragma unroll
  for (int h = 0; h < 2; ++h) {
    const int rbase = m0 + h * 16 + (hi ? 8 : 0);
    #pragma unroll
    for (int nt = 0; nt < 8; ++nt) {
      const int col = (nt << 4) + (lane & 15);
      const float bv = bias ? bias[col] : 0.0f;
      #pragma unroll
      for (int r = 0; r < 8; ++r) {
        const int rowOut = rbase + r;
        if (rowOut < M)
          C[(size_t)rowOut * HD + col] = acc[h * 8 + nt][r] + bv;
      }
    }
  }
}

// ---------------- BatchNorm (training batch stats) + ReLU ----------------
__global__ __launch_bounds__(128) void col_stats(const float* __restrict__ X,
                                                 float* __restrict__ stats, int M) {
  const int c = threadIdx.x;
  float s = 0.f, s2 = 0.f;
  for (int r = blockIdx.x; r < M; r += gridDim.x) {
    const float v = X[(size_t)r * HD + c];
    s += v; s2 += v * v;
  }
  atomicAdd(&stats[c], s);
  atomicAdd(&stats[HD + c], s2);
}

__global__ __launch_bounds__(256) void bn_relu_apply(float* __restrict__ X,
                                                     const float* __restrict__ stats,
                                                     const float* __restrict__ g,
                                                     const float* __restrict__ b, int M) {
  const size_t i = (size_t)blockIdx.x * 256 + threadIdx.x;
  if (i >= (size_t)M * HD) return;
  const int c = (int)(i & (HD - 1));
  const float inv = 1.0f / (float)M;
  const float mu = stats[c] * inv;
  const float var = stats[HD + c] * inv - mu * mu;
  const float v = g[c] * (X[i] - mu) * rsqrtf(var + 1e-5f) + b[c];
  X[i] = v > 0.f ? v : 0.f;
}

// ---------------- GCN degree / symmetric norm ----------------
__global__ __launch_bounds__(256) void deg_count(const int* __restrict__ dst,
                                                 float* __restrict__ deg, int E) {
  const int e = blockIdx.x * 256 + threadIdx.x;
  if (e < E) atomicAdd(&deg[dst[e]], 1.0f);
}

__global__ __launch_bounds__(256) void edge_norm(const int* __restrict__ src,
                                                 const int* __restrict__ dst,
                                                 const float* __restrict__ deg,
                                                 float* __restrict__ norm, int E) {
  const int e = blockIdx.x * 256 + threadIdx.x;
  if (e >= E) return;
  const float a = deg[src[e]], b = deg[dst[e]];
  const float da = a > 0.f ? rsqrtf(a) : 0.f;
  const float db = b > 0.f ? rsqrtf(b) : 0.f;
  norm[e] = da * db;
}

// out[dst] += coef[e] * (den ? 1/den[dst] : 1) * feat[src]  (per channel)
__global__ __launch_bounds__(256) void scatter_edges(const int* __restrict__ src,
                                                     const int* __restrict__ dst,
                                                     const float* __restrict__ coef,
                                                     const float* __restrict__ den,
                                                     const float* __restrict__ feat,
                                                     float* __restrict__ out, int E) {
  const int e = blockIdx.x * 2 + (threadIdx.x >> 7);
  if (e >= E) return;
  const int c = threadIdx.x & 127;
  const int d = dst[e];
  float w = coef[e];
  if (den) w /= den[d];
  atomicAdd(&out[(size_t)d * HD + c], w * feat[(size_t)src[e] * HD + c]);
}

__global__ __launch_bounds__(256) void bias_relu(float* __restrict__ X,
                                                 const float* __restrict__ b, size_t n) {
  const size_t i = (size_t)blockIdx.x * 256 + threadIdx.x;
  if (i >= n) return;
  const float v = X[i] + b[i & (HD - 1)];
  X[i] = v > 0.f ? v : 0.f;
}

// ---------------- GAT pieces ----------------
// dot[row] = sum_c H[row,c]*att[c]; one wave32 per row, butterfly reduce.
__global__ __launch_bounds__(256) void row_dot(const float* __restrict__ H,
                                               const float* __restrict__ att,
                                               float* __restrict__ dot, int M) {
  const int row = blockIdx.x * 8 + (threadIdx.x >> 5);
  const int lane = threadIdx.x & 31;
  if (row >= M) return;
  const float4 h = ((const float4*)(H + (size_t)row * HD))[lane];
  const float4 a = ((const float4*)att)[lane];
  float s = h.x * a.x + h.y * a.y + h.z * a.z + h.w * a.w;
  #pragma unroll
  for (int off = 16; off > 0; off >>= 1) s += __shfl_xor(s, off, 32);
  if (lane == 0) dot[row] = s;
}

__device__ __forceinline__ unsigned fenc(float f) {
  unsigned u = __float_as_uint(f);
  return (u & 0x80000000u) ? ~u : (u | 0x80000000u);
}
__device__ __forceinline__ float fdec(unsigned k) {
  return __uint_as_float((k & 0x80000000u) ? (k ^ 0x80000000u) : ~k);
}

__global__ __launch_bounds__(256) void gat_logit_max(const int* __restrict__ s,
                                                     const int* __restrict__ d,
                                                     const float* __restrict__ ds,
                                                     const float* __restrict__ dd,
                                                     float* __restrict__ ev,
                                                     unsigned* __restrict__ menc, int E) {
  const int e = blockIdx.x * 256 + threadIdx.x;
  if (e >= E) return;
  float x = ds[s[e]] + dd[d[e]];
  x = x > 0.f ? x : 0.2f * x;           // leaky_relu(0.2)
  ev[e] = x;
  atomicMax(&menc[d[e]], fenc(x));
}

__global__ __launch_bounds__(256) void gat_expsum(const int* __restrict__ d,
                                                  float* __restrict__ ev,
                                                  float* __restrict__ den,
                                                  const unsigned* __restrict__ menc, int E) {
  const int e = blockIdx.x * 256 + threadIdx.x;
  if (e >= E) return;
  const float w = __expf(ev[e] - fdec(menc[d[e]]));
  ev[e] = w;
  atomicAdd(&den[d[e]], w);
}

// =====================================================================
// Orchestration
// =====================================================================
extern "C" void kernel_launch(void* const* d_in, const int* in_sizes, int n_in,
                              void* d_out, int out_size, void* d_ws, size_t ws_size,
                              hipStream_t stream) {
  (void)in_sizes; (void)n_in; (void)out_size; (void)ws_size;

  const float* x_gene = (const float*)d_in[0];
  const float* x_mesh = (const float*)d_in[1];
  const int* ppi = (const int*)d_in[2];   // [2, EPP] : row0 src, row1 dst
  const int* mm  = (const int*)d_in[3];
  const int* gme = (const int*)d_in[4];   // gene->mesh
  const int* rgm = (const int*)d_in[5];   // flipped: row0 mesh(src), row1 gene(dst)

  // params, in setup_inputs() dict-insertion order
  enum { P_EMB_G_W = 6, P_EMB_G_B, P_BN_G_G, P_BN_G_B,
         P_EMB_M_W, P_EMB_M_B, P_BN_M_G, P_BN_M_B,
         P_LIN_G_W, P_LIN_G_B, P_LIN_M_W, P_LIN_M_B, P_L0 = 18 };
  const int LSTR = 14;
  const int O_MCONV_W = 0, O_MCONV_B = 1, O_GCONV_W = 2, O_GCONV_B = 3,
            O_GM_WS = 4, O_GM_WD = 5, O_GM_AS = 6, O_GM_AD = 7, O_GM_B = 8,
            O_MG_WS = 9, O_MG_WD = 10, O_MG_AS = 11, O_MG_AD = 12, O_MG_B = 13;
  auto PF = [&](int i) { return (const float*)d_in[i]; };

  // ---- carve workspace
  char* wsb = (char*)d_ws;
  size_t off = 0;
  auto carve = [&](size_t bytes) -> void* {
    void* p = wsb + off;
    off += (bytes + 255) & ~(size_t)255;
    return p;
  };
  float* gx_a = (float*)carve((size_t)NGN * HD * 4);
  float* gx_b = (float*)carve((size_t)NGN * HD * 4);
  float* mx_a = (float*)carve((size_t)NMH * HD * 4);
  float* mx_b = (float*)carve((size_t)NMH * HD * 4);
  float* h1   = (float*)carve((size_t)NGN * HD * 4);   // GAT hs
  float* h2   = (float*)carve((size_t)NGN * HD * 4);   // GAT hd / GCN xw
  float* deg_g    = (float*)carve((size_t)NGN * 4);
  float* deg_m    = (float*)carve((size_t)NMH * 4);
  float* norm_ppi = (float*)carve((size_t)EPP * 4);
  float* norm_mm  = (float*)carve((size_t)EMM * 4);
  float* ev       = (float*)carve((size_t)EGM * 4);
  float* den      = (float*)carve((size_t)NGN * 4);
  unsigned* menc  = (unsigned*)carve((size_t)NGN * 4);
  float* dots     = (float*)carve((size_t)NGN * 4);
  float* dotd     = (float*)carve((size_t)NGN * 4);
  float* stats    = (float*)carve(2 * HD * 4);
  bf16* wp_embg = (bf16*)carve((size_t)60 * 4096 * 2);  // K=1900 -> 60 tiles
  bf16* wp_embm = (bf16*)carve((size_t)24 * 4096 * 2);  // K=768
  auto carve128 = [&]() { return (bf16*)carve((size_t)4 * 4096 * 2); };
  bf16* wp_ling = carve128();
  bf16* wp_linm = carve128();
  bf16 *wp_mconv[2], *wp_gconv[2], *wp_gmws[2], *wp_gmwd[2], *wp_mgws[2], *wp_mgwd[2];
  for (int l = 0; l < 2; ++l) {
    wp_mconv[l] = carve128(); wp_gconv[l] = carve128();
    wp_gmws[l] = carve128();  wp_gmwd[l] = carve128();
    wp_mgws[l] = carve128();  wp_mgwd[l] = carve128();
  }

  auto zero = [&](void* p, size_t bytes) { hipMemsetAsync(p, 0, bytes, stream); };
  auto pack = [&](int pidx, bf16* wp, int K) {
    const size_t total = (size_t)((K + 31) / 32) * 4096;
    pack_w128<<<dim3((unsigned)((total + 255) / 256)), dim3(256), 0, stream>>>(PF(pidx), wp, K);
  };
  auto gemm = [&](const float* A, const bf16* Wp, const float* bias, float* C, int M, int K) {
    gemm_n128<<<dim3((unsigned)((M + 127) / 128)), dim3(128), 0, stream>>>(A, Wp, bias, C, M, K);
  };
  auto relu_bias = [&](float* X, const float* b, int M) {
    const size_t tot = (size_t)M * HD;
    bias_relu<<<dim3((unsigned)((tot + 255) / 256)), dim3(256), 0, stream>>>(X, b, tot);
  };

  // ---- pack all weights to bf16 WMMA fragments
  pack(P_EMB_G_W, wp_embg, KGE);
  pack(P_EMB_M_W, wp_embm, KME);
  pack(P_LIN_G_W, wp_ling, HD);
  pack(P_LIN_M_W, wp_linm, HD);
  for (int l = 0; l < 2; ++l) {
    const int b = P_L0 + l * LSTR;
    pack(b + O_MCONV_W, wp_mconv[l], HD);
    pack(b + O_GCONV_W, wp_gconv[l], HD);
    pack(b + O_GM_WS, wp_gmws[l], HD);
    pack(b + O_GM_WD, wp_gmwd[l], HD);
    pack(b + O_MG_WS, wp_mgws[l], HD);
    pack(b + O_MG_WD, wp_mgwd[l], HD);
  }

  // ---- embeddings + BN(batch stats) + ReLU
  auto bn = [&](float* X, const float* g, const float* b, int M) {
    zero(stats, 2 * HD * 4);
    col_stats<<<dim3(512), dim3(128), 0, stream>>>(X, stats, M);
    const size_t tot = (size_t)M * HD;
    bn_relu_apply<<<dim3((unsigned)((tot + 255) / 256)), dim3(256), 0, stream>>>(X, stats, g, b, M);
  };
  gemm(x_gene, wp_embg, PF(P_EMB_G_B), gx_a, NGN, KGE);
  bn(gx_a, PF(P_BN_G_G), PF(P_BN_G_B), NGN);
  gemm(x_mesh, wp_embm, PF(P_EMB_M_B), mx_a, NMH, KME);
  bn(mx_a, PF(P_BN_M_G), PF(P_BN_M_B), NMH);

  // ---- GCN symmetric norms (once per graph)
  zero(deg_g, (size_t)NGN * 4);
  deg_count<<<dim3((EPP + 255) / 256), dim3(256), 0, stream>>>(ppi + EPP, deg_g, EPP);
  edge_norm<<<dim3((EPP + 255) / 256), dim3(256), 0, stream>>>(ppi, ppi + EPP, deg_g, norm_ppi, EPP);
  zero(deg_m, (size_t)NMH * 4);
  deg_count<<<dim3((EMM + 255) / 256), dim3(256), 0, stream>>>(mm + EMM, deg_m, EMM);
  edge_norm<<<dim3((EMM + 255) / 256), dim3(256), 0, stream>>>(mm, mm + EMM, deg_m, norm_mm, EMM);

  // ---- GCN stacks
  auto gcn = [&](float*& xin, float*& xout, const int* src, const int* dst,
                 const float* norm, int E, const bf16* Wp, const float* bias, int n) {
    gemm(xin, Wp, nullptr, h2, n, HD);
    zero(xout, (size_t)n * HD * 4);
    scatter_edges<<<dim3((unsigned)((E + 1) / 2)), dim3(256), 0, stream>>>(
        src, dst, norm, nullptr, h2, xout, E);
    relu_bias(xout, bias, n);
    float* t = xin; xin = xout; xout = t;
  };
  for (int l = 0; l < 2; ++l)
    gcn(mx_a, mx_b, mm, mm + EMM, norm_mm, EMM, wp_mconv[l], PF(P_L0 + l * LSTR + O_MCONV_B), NMH);
  for (int l = 0; l < 2; ++l)
    gcn(gx_a, gx_b, ppi, ppi + EPP, norm_ppi, EPP, wp_gconv[l], PF(P_L0 + l * LSTR + O_GCONV_B), NGN);

  // ---- bipartite GAT cross-updates
  auto gat = [&](const float* xsrc, int Ns, const float* xdst, int Nd,
                 const int* s, const int* d, int E,
                 const bf16* Ws, const bf16* Wd, const float* as_, const float* ad_,
                 const float* bias, float* out) {
    gemm(xsrc, Ws, nullptr, h1, Ns, HD);
    gemm(xdst, Wd, nullptr, h2, Nd, HD);
    row_dot<<<dim3((unsigned)((Ns + 7) / 8)), dim3(256), 0, stream>>>(h1, as_, dots, Ns);
    row_dot<<<dim3((unsigned)((Nd + 7) / 8)), dim3(256), 0, stream>>>(h2, ad_, dotd, Nd);
    zero(menc, (size_t)Nd * 4);  // 0 <= fenc(x) for all finite x
    gat_logit_max<<<dim3((E + 255) / 256), dim3(256), 0, stream>>>(s, d, dots, dotd, ev, menc, E);
    zero(den, (size_t)Nd * 4);
    gat_expsum<<<dim3((E + 255) / 256), dim3(256), 0, stream>>>(d, ev, den, menc, E);
    zero(out, (size_t)Nd * HD * 4);
    scatter_edges<<<dim3((unsigned)((E + 1) / 2)), dim3(256), 0, stream>>>(s, d, ev, den, h1, out, E);
    relu_bias(out, bias, Nd);
  };
  for (int l = 0; l < 2; ++l) {
    const int b = P_L0 + l * LSTR;
    // new mesh features from (gene -> mesh)
    gat(gx_a, NGN, mx_a, NMH, gme, gme + EGM, EGM,
        wp_gmws[l], wp_gmwd[l], PF(b + O_GM_AS), PF(b + O_GM_AD), PF(b + O_GM_B), mx_b);
    // new gene features from (mesh -> gene), using OLD mx_a
    gat(mx_a, NMH, gx_a, NGN, rgm, rgm + EGM, EGM,
        wp_mgws[l], wp_mgwd[l], PF(b + O_MG_AS), PF(b + O_MG_AD), PF(b + O_MG_B), gx_b);
    float* t;
    t = gx_a; gx_a = gx_b; gx_b = t;
    t = mx_a; mx_a = mx_b; mx_b = t;
  }

  // ---- final linears -> d_out (gene block then mesh block)
  float* out_g = (float*)d_out;
  float* out_m = out_g + (size_t)NGN * HD;
  gemm(gx_a, wp_ling, PF(P_LIN_G_B), out_g, NGN, HD);
  gemm(mx_a, wp_linm, PF(P_LIN_M_B), out_m, NMH, HD);
}
